// Kernel_17371628126399981364_77051713290582
// MI455X (gfx1250) — matrix operand order repaired and checked by reading, not compiled, not run on hardware
//
#include <hip/hip_runtime.h>
#include <hip/hip_bf16.h>
#include <stdint.h>

#define NB 32
#define CH 64
#define HH 96
#define WW 96
#define HW (HH*WW)            // 9216

typedef __attribute__((ext_vector_type(16))) __bf16 v16bf;
typedef __attribute__((ext_vector_type(8)))  float  v8f;

union Frag { v16bf v; uint32_t u[8]; uint4 q[2]; };

__device__ __forceinline__ uint32_t pk_bf16(float a, float b) {
    uint32_t ua = __float_as_uint(a), ub = __float_as_uint(b);
    uint32_t ra = (ua + 0x7FFFu + ((ua >> 16) & 1u)) >> 16;
    uint32_t rb = (ub + 0x7FFFu + ((ub >> 16) & 1u)) >> 16;
    return (ra & 0xFFFFu) | (rb << 16);
}
__device__ __forceinline__ unsigned short f2bf(float a) {
    uint32_t u = __float_as_uint(a);
    return (unsigned short)((u + 0x7FFFu + ((u >> 16) & 1u)) >> 16);
}

// ---------------------------------------------------------------------------
// K0a: pack x -> xT (bf16 channel pairs, [n][s][c/2] dwords). WMMA-B layout
// for channel-K GEMMs (k_main, k_p6mm). LDS tile keeps both sides coalesced.
// ---------------------------------------------------------------------------
__global__ __launch_bounds__(256) void k_pack(const float* __restrict__ x,
                                              uint32_t* __restrict__ xT)
{
    __shared__ float tile[CH][65];
    const int n  = blockIdx.y;
    const int sb = blockIdx.x * 64;          // 144 s-blocks
    const int tid = threadIdx.x;
    const float* xn = x + (size_t)n * CH * HW;
    const int sl = tid & 63, c0 = tid >> 6;
#pragma unroll
    for (int i = 0; i < 16; ++i) {
        int c = i * 4 + c0;
        tile[c][sl] = xn[(size_t)c * HW + sb + sl];
    }
    __syncthreads();
    const int cp = tid & 31, s0 = tid >> 5;
    uint32_t* outn = xT + ((size_t)n * HW + sb) * 32;
#pragma unroll
    for (int j = 0; j < 8; ++j) {
        int s = j * 8 + s0;
        outn[(size_t)s * 32 + cp] = pk_bf16(tile[2 * cp][s], tile[2 * cp + 1][s]);
    }
}

// ---------------------------------------------------------------------------
// K0b: pack rolled x -> xRr (bf16 spatial pairs, [n][c][s/2] dwords).
// xRr[n,c,s] = x[n,c,(h-1)%96,(w+1)%96]. WMMA A/B layout for spatial-K GEMMs.
// ---------------------------------------------------------------------------
__global__ __launch_bounds__(256) void k_packroll(const float* __restrict__ x,
                                                  uint32_t* __restrict__ xRr)
{
    const int row = blockIdx.x;              // n*CH + c, 2048 rows
    const float* base = x + (size_t)row * HW;
    uint32_t* outr = xRr + (size_t)row * (HW / 2);
    for (int jd = threadIdx.x; jd < HW / 2; jd += 256) {
        int s0 = 2 * jd;
        int h = s0 / WW, w = s0 - h * WW;
        int hs = (h + HH - 1) % HH;
        int src0 = hs * WW + (w + 1) % WW;
        int src1 = hs * WW + (w + 2) % WW;
        outr[jd] = pk_bf16(base[src0], base[src1]);
    }
}

// ---------------------------------------------------------------------------
// K1: t2 partials: t2p[q][n,c,d] = sum_{k in chunk q} x[n,c,k]*x_roll[n,d,k]
// A packed on the fly from contiguous fp32 pairs; B = 2x b128 from xRr.
// 4-way K split -> 512 waves. Partials summed (deterministically) in k_t8.
// ---------------------------------------------------------------------------
__global__ __launch_bounds__(32) void k_gram(const float* __restrict__ x,
                                             const uint32_t* __restrict__ xRr,
                                             float* __restrict__ t2p)
{
    const int lane = threadIdx.x;
    const int bid  = blockIdx.x;          // 0..511
    const int q  = bid & 3, ct = (bid >> 2) & 3, n = bid >> 4;
    const int m  = lane & 15, hi = lane >> 4;

    const float* arow = x + ((size_t)n * CH + ct * 16 + m) * HW;
    const uint32_t* xRn = xRr + (size_t)n * CH * (HW / 2);

    v8f acc[4];
#pragma unroll
    for (int i = 0; i < 4; ++i)
#pragma unroll
        for (int r = 0; r < 8; ++r) acc[i][r] = 0.0f;

    const int kbeg = q * (HW / 4), kend = kbeg + (HW / 4);
    for (int kb = kbeg; kb < kend; kb += 32) {
        Frag a;
#pragma unroll
        for (int v = 0; v < 8; ++v) {
            int kidx = kb + hi * 16 + v * 2;
            a.u[v] = pk_bf16(arow[kidx], arow[kidx + 1]);
        }
#pragma unroll
        for (int dt = 0; dt < 4; ++dt) {
            const uint4* bp = (const uint4*)(xRn + (size_t)(dt * 16 + m) * (HW / 2)
                                             + (kb >> 1) + hi * 8);
            Frag b;
            b.q[0] = bp[0];
            b.q[1] = bp[1];
            acc[dt] = __builtin_amdgcn_wmma_f32_16x16x32_bf16(
                false, a.v, false, b.v, (short)0, acc[dt], false, false);
        }
    }
    float* t2n = t2p + ((size_t)q * NB + n) * CH * CH;
#pragma unroll
    for (int dt = 0; dt < 4; ++dt)
#pragma unroll
        for (int r = 0; r < 8; ++r) {
            int c = ct * 16 + hi * 8 + r;
            int d = dt * 16 + m;
            t2n[c * CH + d] = acc[dt][r];
        }
}

// ---------------------------------------------------------------------------
// K2: A_kk[n,c,b] = (1/sqrt(448)) * sum_bb (p5[bb,c]*t2[n,bb,c]/96) * p8[bb,b*7+kk]
// t2 = sum of 4 partials. Output bf16, b contiguous (A-fragment dword pairs).
// ---------------------------------------------------------------------------
__global__ __launch_bounds__(256) void k_t8(const float* __restrict__ t2p,
                                            const float* __restrict__ p5,
                                            const float* __restrict__ p8,
                                            unsigned short* __restrict__ Abf)
{
    __shared__ float t5s[CH][CH];   // [bb][c]
    __shared__ float p8s[CH][CH];   // [bb][bch]
    const int kk = blockIdx.x;      // 0..6
    const int n  = blockIdx.y;
    const int tid = threadIdx.x;
    const size_t nb = (size_t)n * CH * CH, qs = (size_t)NB * CH * CH;
    for (int i = tid; i < CH * CH; i += 256) {
        int bb = i >> 6, c = i & 63;
        float t2v = t2p[nb + i] + t2p[qs + nb + i] + t2p[2 * qs + nb + i] + t2p[3 * qs + nb + i];
        t5s[bb][c] = p5[i] * t2v * (1.0f / 96.0f);
        p8s[bb][c] = p8[bb * (CH * 7) + c * 7 + kk];
    }
    __syncthreads();
    unsigned short* An = Abf + (((size_t)n * 7 + kk) * CH * CH);
    const float sc = 0.0472455591f;  // 1/sqrt(448)
    for (int i = tid; i < CH * CH; i += 256) {
        int c = i >> 6, bch = i & 63;
        float s = 0.0f;
#pragma unroll
        for (int bb = 0; bb < CH; ++bb) s += t5s[bb][c] * p8s[bb][bch];
        An[c * CH + bch] = f2bf(s * sc);
    }
}

// ---------------------------------------------------------------------------
// K3: P[n,j,s'] = sum_c (w4[c,j]*p6[c]) * t1[n,c,s']  (M=16 pad of 9 taps,
// K=64 channels, N=spatial). B = xT at the ROLLED spatial index (channel
// packing is roll-invariant). One wave per s-tile, 2 WMMAs.
// ---------------------------------------------------------------------------
__global__ __launch_bounds__(256) void k_p6mm(const uint32_t* __restrict__ xT,
                                              const float* __restrict__ w4,
                                              const float* __restrict__ p6,
                                              float* __restrict__ P)
{
    const int n    = blockIdx.y;
    const int tid  = threadIdx.x;
    const int wave = tid >> 5, lane = tid & 31;
    const int stile = blockIdx.x * 8 + wave;
    const int s0 = stile * 16;
    const int h  = s0 / WW;
    const int wb = s0 - h * WW;
    const int nn = lane & 15, hi = lane >> 4;

    // A fragments (j rows, channel K), loop invariant
    Frag a[2];
#pragma unroll
    for (int ks = 0; ks < 2; ++ks)
#pragma unroll
        for (int v = 0; v < 8; ++v) {
            int c0 = ks * 32 + hi * 16 + v * 2;
            float f0 = (nn < 9) ? w4[c0 * 9 + nn] * p6[c0] : 0.0f;
            float f1 = (nn < 9) ? w4[(c0 + 1) * 9 + nn] * p6[c0 + 1] : 0.0f;
            a[ks].u[v] = pk_bf16(f0, f1);
        }

    // B: xT at rolled source of s' = s0+nn
    int hs = (h + HH - 1) % HH;
    int wsrc = (wb + nn + 1) % WW;
    const uint4* bp = (const uint4*)(xT + ((size_t)n * HW + hs * WW + wsrc) * 32 + hi * 8);

    v8f acc;
#pragma unroll
    for (int r = 0; r < 8; ++r) acc[r] = 0.0f;
#pragma unroll
    for (int ks = 0; ks < 2; ++ks) {
        Frag b;
        b.q[0] = bp[ks * 4];
        b.q[1] = bp[ks * 4 + 1];
        acc = __builtin_amdgcn_wmma_f32_16x16x32_bf16(
            false, a[ks].v, false, b.v, (short)0, acc, false, false);
    }
#pragma unroll
    for (int r = 0; r < 8; ++r) {
        int j = r + hi * 8;
        if (j < 9) P[((size_t)n * 9 + j) * HW + s0 + nn] = acc[r];
    }
}

// ---------------------------------------------------------------------------
// K4: t6[n,s] = sum_{taps in range} P[n, j, shifted s]. Also zero-inits R8.
// ---------------------------------------------------------------------------
__global__ __launch_bounds__(256) void k_t6g(const float* __restrict__ P,
                                             float* __restrict__ t6,
                                             float* __restrict__ R8)
{
    const int n = blockIdx.y;
    const int s = blockIdx.x * 256 + threadIdx.x;
    const int h = s / WW, w = s - h * WW;
    const float* Pn = P + (size_t)n * 9 * HW;
    float acc = 0.0f;
#pragma unroll
    for (int i = 0; i < 3; ++i) {
        int hh = h + 2 * (i - 1);
        if (hh < 0 || hh >= HH) continue;
#pragma unroll
        for (int j = 0; j < 3; ++j) {
            int ww = w + 2 * (j - 1);
            if (ww < 0 || ww >= WW) continue;
            acc += Pn[(size_t)(i * 3 + j) * HW + hh * WW + ww];
        }
    }
    t6[(size_t)n * HW + s] = acc;
    // zero R8 (8*NB*CH*32 = 524288 floats); 294912 threads -> 2 each
    int g = n * HW + s;
    R8[g] = 0.0f;
    int g2 = g + NB * HW;
    if (g2 < 8 * NB * CH * 32) R8[g2] = 0.0f;
}

// ---------------------------------------------------------------------------
// K5: R8[q][n,c,j2] = sum_{s' in chunk q} t1[n,c,s'] * u[n,s',j2]
// u[s',(a,b)] = t6[h'-a, w'-b] (0 outside). A = xRr (spatial-K pairs),
// B built from bounds-checked t6 loads. M=64 (4 tiles) x N=32 (25 used).
// ---------------------------------------------------------------------------
__global__ __launch_bounds__(32) void k_t9mm(const uint32_t* __restrict__ xRr,
                                             const float* __restrict__ t6,
                                             float* __restrict__ R8)
{
    const int q = blockIdx.x;     // 0..7 K-split
    const int n = blockIdx.y;
    const int lane = threadIdx.x;
    const int nn = lane & 15, hi = lane >> 4;

    const uint32_t* xRn = xRr + (size_t)n * CH * (HW / 2);
    const float* t6n = t6 + (size_t)n * HW;

    v8f acc[4][2];
#pragma unroll
    for (int i = 0; i < 4; ++i)
#pragma unroll
        for (int j = 0; j < 2; ++j)
#pragma unroll
            for (int r = 0; r < 8; ++r) acc[i][j][r] = 0.0f;

    const int kbeg = q * (HW / 8), kend = kbeg + (HW / 8);
    for (int kb = kbeg; kb < kend; kb += 32) {
        Frag b[2];
#pragma unroll
        for (int jt = 0; jt < 2; ++jt) {
            int j2 = jt * 16 + nn;
            bool vc = j2 < 25;
            int a = (vc ? j2 / 5 : 0) - 2;
            int bb = (vc ? j2 % 5 : 0) - 2;
#pragma unroll
            for (int v = 0; v < 8; ++v) {
                int sp = kb + ((v < 4) ? 0 : 16) + hi * 8 + (v & 3) * 2;       // even, pair stays in row
                int hp = sp / WW, w0 = sp - hp * WW;
                int hh = hp - a;
                bool ih = (hh >= 0) && (hh < HH) && vc;
                int wa = w0 - bb, wbv = wa + 1;
                float f0 = (ih && wa >= 0 && wa < WW) ? t6n[hh * WW + wa] : 0.0f;
                float f1 = (ih && wbv >= 0 && wbv < WW) ? t6n[hh * WW + wbv] : 0.0f;
                b[jt].u[v] = pk_bf16(f0, f1);
            }
        }
#pragma unroll
        for (int ct = 0; ct < 4; ++ct) {
            const uint4* pa = (const uint4*)(xRn + (size_t)(ct * 16 + nn) * (HW / 2)
                                             + (kb >> 1) + hi * 4);
            Frag af;
            af.q[0] = pa[0];
            af.q[1] = pa[2];   // +8 dwords
#pragma unroll
            for (int jt = 0; jt < 2; ++jt)
                acc[ct][jt] = __builtin_amdgcn_wmma_f32_16x16x32_bf16(
                    false, af.v, false, b[jt].v, (short)0, acc[ct][jt], false, false);
        }
    }
    float* Rq = R8 + (((size_t)q * NB + n) * CH) * 32;
#pragma unroll
    for (int ct = 0; ct < 4; ++ct)
#pragma unroll
        for (int jt = 0; jt < 2; ++jt)
#pragma unroll
            for (int r = 0; r < 8; ++r) {
                int c = ct * 16 + hi * 8 + r;
                Rq[(size_t)c * 32 + jt * 16 + nn] = acc[ct][jt][r];
            }
}

// ---------------------------------------------------------------------------
// K6: t9[n,c] = (1/96) * sum_{j2<25} w3[c,j2] * sum_q R8[q][n,c,j2]
// ---------------------------------------------------------------------------
__global__ __launch_bounds__(64) void k_t9fin(const float* __restrict__ R8,
                                              const float* __restrict__ w3,
                                              float* __restrict__ t9)
{
    const int n = blockIdx.x, c = threadIdx.x;
    float s = 0.0f;
    for (int j2 = 0; j2 < 25; ++j2) {
        float r = 0.0f;
#pragma unroll
        for (int q = 0; q < 8; ++q)
            r += R8[(((size_t)q * NB + n) * CH + c) * 32 + j2];
        s += w3[c * 25 + j2] * r;
    }
    t9[n * CH + c] = s * (1.0f / 96.0f);
}

// ---------------------------------------------------------------------------
// K7 (main WMMA): out[n,c,s] = t9[n,c] + sum_{kk,b} A_kk[n,c,b]*x[n,b,h,w+2kk-6]
// A matrices (57KB) in LDS (ds_load_b128 fragments); B = 2x b128 from xT.
// Each wave: 4 c-tiles x 1 s-tile, 7 kk x 2 ksteps x 4 WMMAs = 56 WMMA.
// ---------------------------------------------------------------------------
__global__ __launch_bounds__(256) void k_main(const uint32_t* __restrict__ xT,
                                              const uint32_t* __restrict__ Apk,
                                              const float* __restrict__ t9,
                                              float* __restrict__ out)
{
    __shared__ __align__(16) uint32_t As[7 * CH * 32];
    const int n   = blockIdx.y;
    const int tid = threadIdx.x;
    const uint32_t* An = Apk + (size_t)n * (7 * CH * 32);
    for (int i = tid; i < 7 * CH * 32; i += 256) As[i] = An[i];
    __syncthreads();

    const int wave = tid >> 5, lane = tid & 31;
    const int stile = blockIdx.x * 8 + wave;
    const int s0 = stile * 16;
    const int h  = s0 / WW;
    const int wb = s0 - h * WW;
    const int nn = lane & 15, hi = lane >> 4;

    const uint32_t* xTn = xT + (size_t)n * HW * 32;

    v8f acc[4];
#pragma unroll
    for (int i = 0; i < 4; ++i)
#pragma unroll
        for (int r = 0; r < 8; ++r) acc[i][r] = 0.0f;

    for (int kk = 0; kk < 7; ++kk) {
        int wsh = wb + nn + 2 * kk - 6;
        bool valid = (wsh >= 0) && (wsh < WW);
        int wclamp = valid ? wsh : 0;
        const uint4* bp = (const uint4*)(xTn + (size_t)(h * WW + wclamp) * 32 + hi * 8);
#pragma unroll
        for (int kstep = 0; kstep < 2; ++kstep) {
            Frag bf;
            uint4 q0 = bp[kstep * 4];
            uint4 q1 = bp[kstep * 4 + 1];
            if (!valid) { q0.x = q0.y = q0.z = q0.w = 0u; q1.x = q1.y = q1.z = q1.w = 0u; }
            bf.q[0] = q0; bf.q[1] = q1;
#pragma unroll
            for (int ct = 0; ct < 4; ++ct) {
                int row = (kk * CH + ct * 16 + nn) * 32;
                const uint4* pa = (const uint4*)&As[row + kstep * 16 + hi * 8];
                Frag af;
                af.q[0] = pa[0];
                af.q[1] = pa[1];
                acc[ct] = __builtin_amdgcn_wmma_f32_16x16x32_bf16(
                    false, af.v, false, bf.v, (short)0, acc[ct], false, false);
            }
        }
    }

#pragma unroll
    for (int ct = 0; ct < 4; ++ct) {
        const float* t9p = t9 + n * CH + ct * 16 + hi * 8;
        float4 ta = *(const float4*)t9p;
        float4 tb = *(const float4*)(t9p + 4);
        float tv[8] = {ta.x, ta.y, ta.z, ta.w, tb.x, tb.y, tb.z, tb.w};
#pragma unroll
        for (int r = 0; r < 8; ++r) {
            int c = ct * 16 + hi * 8 + r;
            out[((size_t)n * CH + c) * HW + s0 + nn] = acc[ct][r] + tv[r];
        }
    }
}

extern "C" void kernel_launch(void* const* d_in, const int* in_sizes, int n_in,
                              void* d_out, int out_size, void* d_ws, size_t ws_size,
                              hipStream_t stream)
{
    const float* x  = (const float*)d_in[0];
    const float* w3 = (const float*)d_in[1];
    const float* w4 = (const float*)d_in[2];
    const float* p5 = (const float*)d_in[3];
    const float* p6 = (const float*)d_in[4];
    const float* p8 = (const float*)d_in[5];
    float* out = (float*)d_out;

    char* ws = (char*)d_ws;
    float*          t2p = (float*)(ws + 0);                  // 2 MB (4 partials)
    unsigned short* Abf = (unsigned short*)(ws + 2097152);   // 1.75 MB
    float*          t6  = (float*)(ws + 3932160);            // 1.125 MB
    float*          t9  = (float*)(ws + 5111808);            // 8 KB
    float*          P   = (float*)(ws + 5120000);            // 10.125 MB
    float*          R8  = (float*)(ws + 15736832);           // 2 MB (8 partials)
    uint32_t*       xT  = (uint32_t*)(ws + 17833984);        // 36 MB
    uint32_t*       xRr = (uint32_t*)(ws + 55582720);        // 36 MB

    k_pack    <<<dim3(144, 32), dim3(256), 0, stream>>>(x, xT);
    k_packroll<<<dim3(2048),    dim3(256), 0, stream>>>(x, xRr);
    k_gram    <<<dim3(512),     dim3(32),  0, stream>>>(x, xRr, t2p);
    k_t8      <<<dim3(7, 32),   dim3(256), 0, stream>>>(t2p, p5, p8, Abf);
    k_p6mm    <<<dim3(72, 32),  dim3(256), 0, stream>>>(xT, w4, p6, P);
    k_t6g     <<<dim3(36, 32),  dim3(256), 0, stream>>>(P, t6, R8);
    k_t9mm    <<<dim3(8, 32),   dim3(32),  0, stream>>>(xRr, t6, R8);
    k_t9fin   <<<dim3(32),      dim3(64),  0, stream>>>(R8, w3, t9);
    k_main    <<<dim3(72, 32),  dim3(256), 0, stream>>>(xT, (const uint32_t*)Abf, t9, out);
}
